// ProcessModelsampling_36876589203559
// MI455X (gfx1250) — matrix operand order repaired and checked by reading, not compiled, not run on hardware
//
#include <hip/hip_runtime.h>
#include <hip/hip_bf16.h>

// Fused flipout-MLP forward for MI455X (gfx1250, wave32, WMMA).
// 512 blocks x 128 threads (4 waves). Each wave carries a 16-row tile through
// all 10 layers using v_wmma_f32_16x16x32_f16. Activations live in LDS;
// per-layer weights are staged into LDS in B-fragment register order.
// Round 2: forceinline + const-propagated dims (full unroll, no v_movrels),
// single-round sign hash (2 bits per hash), hoisted PE row index.

typedef __attribute__((ext_vector_type(16))) _Float16 v16h;
typedef __attribute__((ext_vector_type(8)))  _Float16 v8h;
typedef __attribute__((ext_vector_type(8)))  float    v8f;

#define N_ROWS  32768
#define ROWS_PB 64
#define THREADS 128

struct Params {
  const float* last_state;   // [N,7]
  const float* action;       // [N,7]
  const float* sample_freq;  // [N]
  const float* pe;           // [4,128]
  const float* fw[8];        // flipout mw:  b1,b2,b3,a1,a2,a3,b4,b5
  const float* fr[8];        // flipout rw
  const float* fmb[8];       // flipout mb
  const float* frb[8];       // flipout rb
  const float* m1w; const float* m1b;   // [128,128],[128]
  const float* m2w; const float* m2b;   // [64,128],[64]
  float* out_state;          // [N,7]
  float* out_feat;           // [N,64]
};

// ---------------- RNG stand-ins (JAX threefry not bit-reproducible) ---------
// Heavy hash + gaussian: used only in per-layer weight staging (amortized).
__device__ __forceinline__ unsigned pcg(unsigned x) {
  x = x * 747796405u + 2891336453u;
  unsigned w = ((x >> ((x >> 28) + 4u)) ^ x) * 277803737u;
  return (w >> 22) ^ w;
}
__device__ __forceinline__ float gaussian(unsigned h) {
  float u1 = (float)((h >> 8) & 0xFFFFFFu) * (1.f / 16777216.f) + 1e-7f;
  unsigned h2 = pcg(h ^ 0x85EBCA6Bu);
  float u2 = (float)((h2 >> 8) & 0xFFFFFFu) * (1.f / 16777216.f);
  return sqrtf(-2.f * logf(u1)) * cosf(6.28318530718f * u2);
}
// Cheap single-round mix for per-element signs (each sign consumed once, at
// write time, so no cross-site consistency is required). Two sign bits/hash.
__device__ __forceinline__ unsigned mix1(unsigned seed, unsigned a, unsigned b) {
  unsigned x = seed ^ (a * 0x9E3779B9u) ^ (b * 0x85EBCA6Bu);
  x ^= x >> 16;
  x *= 0x7FEB352Du;
  return x ^ (x >> 15);
}
__device__ __forceinline__ float bitsgn(unsigned h, int bit) {
  return ((h >> bit) & 1u) ? 1.f : -1.f;
}

// ---------------- fragment loads ----------------
// A (16x32 f16): lanes 0-15 = M, lane>>4 picks K-halves {0-7,16-23} vs {8-15,24-31}
__device__ __forceinline__ v16h loadA(const _Float16* buf, int ld, int row0, int kt, int lane) {
  int m = lane & 15, kh = lane >> 4;
  const _Float16* p = buf + (row0 + m) * ld + kt * 32 + kh * 8;
  union { v16h v; v8h h[2]; } u;
  u.h[0] = *(const v8h*)p;         // K = kt*32 + kh*8 .. +7
  u.h[1] = *(const v8h*)(p + 16);  // K = kt*32 + 16 + kh*8 .. +7
  return u.v;
}
// B fragments pre-swizzled: per (kt,nt) fragment, lane's 16 halves contiguous.
__device__ __forceinline__ v16h loadB(const _Float16* W, int ntiles, int kt, int nt, int lane) {
  const _Float16* p = W + (kt * ntiles + nt) * 512 + lane * 16;
  union { v16h v; v8h h[2]; } u;
  u.h[0] = *(const v8h*)p;
  u.h[1] = *(const v8h*)(p + 8);
  return u.v;
}

// ---------------- weight staging: global fp32 -> LDS f16 B-fragment layout ---
__device__ __forceinline__ void stage_weights(
    const float* mw, const float* rw, const float* mb, const float* rb,
    int din, int dout, bool flip, unsigned seed,
    _Float16* WME, _Float16* WPE, float* BM, float* BP) {
  const int ktiles = (din + 31) >> 5, ntiles = (dout + 15) >> 4;
  const int total = ktiles * ntiles * 512;
  for (int i = threadIdx.x; i < total; i += THREADS) {
    int e = i & 15, lane = (i >> 4) & 31, f = i >> 9;
    int nt = f % ntiles, kt = f / ntiles;
    int n = nt * 16 + (lane & 15);
    int k = kt * 32 + ((lane >> 4) << 3) + (e & 7) + ((e >> 3) << 4);
    bool ok = (k < din) && (n < dout);
    float m = ok ? mw[n * din + k] : 0.f;
    WME[f * 512 + lane * 16 + e] = (_Float16)m;
    if (flip) {
      float p = 0.f;
      if (ok) {
        float sp = log1pf(expf(rw[n * din + k]));  // softplus(rho)
        p = sp * gaussian(pcg(seed ^ (unsigned)(n * 131 + k)));
      }
      WPE[f * 512 + lane * 16 + e] = (_Float16)p;
    }
  }
  for (int i = threadIdx.x; i < ntiles * 16; i += THREADS) {
    BM[i] = (i < dout) ? mb[i] : 0.f;
    float p = 0.f;
    if (flip && i < dout) p = log1pf(expf(rb[i])) * gaussian(pcg(seed ^ 0x5BD1u ^ (unsigned)i));
    BP[i] = p;
  }
}

// ---------------- one flipout/dense layer for this wave's 16-row tile --------
__device__ __forceinline__ void run_layer(
    const _Float16* in, const _Float16* ins, int ldin,
    _Float16* out, _Float16* outs, int ldout, int ncol0,
    const _Float16* WME, const _Float16* WPE,
    const float* BM, const float* BP,
    int din, int dout, bool flip, bool relu,
    unsigned sout_seed, unsigned nsin_seed,
    int grow0, int trow, float* feat, int lane) {
  const int ktiles = (din + 31) >> 5, ntiles = (dout + 15) >> 4;
#pragma unroll
  for (int nt = 0; nt < ntiles; ++nt) {
    v8f acc = {};
#pragma unroll
    for (int kt = 0; kt < ktiles; ++kt) {
      v16h a = loadA(in, ldin, trow, kt, lane);
      v16h b = loadB(WME, ntiles, kt, nt, lane);
      acc = __builtin_amdgcn_wmma_f32_16x16x32_f16(false, a, false, b, (short)0, acc, false, false);
    }
    v8f acc2 = {};
    if (flip) {
#pragma unroll
      for (int kt = 0; kt < ktiles; ++kt) {
        v16h a = loadA(ins, ldin, trow, kt, lane);
        v16h b = loadB(WPE, ntiles, kt, nt, lane);
        acc2 = __builtin_amdgcn_wmma_f32_16x16x32_f16(false, a, false, b, (short)0, acc2, false, false);
      }
    }
    // C layout: lanes 0-15 N=lane, M=r; lanes 16-31 N=lane-16, M=8+r
    const int n = nt * 16 + (lane & 15);
    const int mh = (lane >> 4) * 8;
#pragma unroll
    for (int r = 0; r < 8; ++r) {
      int mrow = mh + r;
      int grow = grow0 + mrow;
      unsigned h = mix1(sout_seed ^ (nsin_seed << 8), (unsigned)grow, (unsigned)(ncol0 + n));
      float v = acc[r] + BM[n];
      if (flip) v += (acc2[r] + BP[n]) * bitsgn(h, 0);
      if (relu) v = fmaxf(v, 0.f);
      out[(trow + mrow) * ldout + ncol0 + n] = (_Float16)v;
      if (outs)
        outs[(trow + mrow) * ldout + ncol0 + n] = (_Float16)(v * bitsgn(h, 1));
      if (feat) feat[grow * dout + n] = v;
    }
  }
}

// ---------------- stage a 7-wide fp32 input into a padded 32-wide f16 buffer -
__device__ __forceinline__ void stage_input(const float* src, _Float16* dst, _Float16* dsts,
                                            unsigned sin_seed, int grow_base) {
  for (int i = threadIdx.x; i < ROWS_PB * 32; i += THREADS) {
    int r = i >> 5, c = i & 31;
    int grow = grow_base + r;
    float v = (c < 7) ? src[grow * 7 + c] : 0.f;
    dst[r * 32 + c] = (_Float16)v;
    dsts[r * 32 + c] = (_Float16)(v * bitsgn(mix1(sin_seed, (unsigned)grow, (unsigned)c), 0));
  }
}

__global__ void __launch_bounds__(THREADS)
flipout_mlp_kernel(Params P) {
  extern __shared__ __align__(16) char smem_raw[];
  _Float16* B0  = (_Float16*)smem_raw;        // 64x128
  _Float16* B1  = B0 + 8192;                  // 64x128
  _Float16* B2  = B1 + 8192;                  // 64x128
  _Float16* S0  = B2 + 8192;                  // signed ping
  _Float16* S1  = S0 + 8192;                  // signed pong
  _Float16* WME = S1 + 8192;                  // 16384 halves (max 128x128)
  _Float16* WPE = WME + 16384;                // 8192 halves (max 128x64 flip)
  float* BM = (float*)(WPE + 8192);           // 128
  float* BP = BM + 128;                       // 128

  const int lane = threadIdx.x & 31;
  const int wave = threadIdx.x >> 5;
  const int trow = wave * 16;
  const int grow0 = blockIdx.x * ROWS_PB + trow;
  const int gbase = blockIdx.x * ROWS_PB;

  // seeds per flipout layer l: w=+0, s_in=+2, s_out=+3
  auto SD = [](int l, int which) { return (unsigned)((l + 1) * 0x1000 + which); };

  // ---- state branch: x -> b1 -> b2 -> b3 (relu) ----
  stage_input(P.last_state, B0, S0, SD(0, 2), gbase);
  stage_weights(P.fw[0], P.fr[0], P.fmb[0], P.frb[0], 7, 64, true, SD(0, 0), WME, WPE, BM, BP);
  __syncthreads();
  run_layer(B0, S0, 32, B1, S1, 64, 0, WME, WPE, BM, BP, 7, 64, true, true,
            SD(0, 3), SD(1, 2), grow0, trow, nullptr, lane);
  __syncthreads();
  stage_weights(P.fw[1], P.fr[1], P.fmb[1], P.frb[1], 64, 128, true, SD(1, 0), WME, WPE, BM, BP);
  __syncthreads();
  run_layer(B1, S1, 64, B0, S0, 128, 0, WME, WPE, BM, BP, 64, 128, true, true,
            SD(1, 3), SD(2, 2), grow0, trow, nullptr, lane);
  __syncthreads();
  stage_weights(P.fw[2], P.fr[2], P.fmb[2], P.frb[2], 128, 64, true, SD(2, 0), WME, WPE, BM, BP);
  __syncthreads();
  run_layer(B0, S0, 128, B1, nullptr, 128, 0, WME, WPE, BM, BP, 128, 64, true, true,
            SD(2, 3), 0u, grow0, trow, nullptr, lane);   // -> merge cols 0..63 in B1
  __syncthreads();

  // ---- action branch: a -> a1 -> a2 -> a3 (relu) ----
  stage_input(P.action, B0, S0, SD(3, 2), gbase);
  stage_weights(P.fw[3], P.fr[3], P.fmb[3], P.frb[3], 7, 64, true, SD(3, 0), WME, WPE, BM, BP);
  __syncthreads();
  run_layer(B0, S0, 32, B2, S1, 64, 0, WME, WPE, BM, BP, 7, 64, true, true,
            SD(3, 3), SD(4, 2), grow0, trow, nullptr, lane);
  __syncthreads();
  stage_weights(P.fw[4], P.fr[4], P.fmb[4], P.frb[4], 64, 128, true, SD(4, 0), WME, WPE, BM, BP);
  __syncthreads();
  run_layer(B2, S1, 64, B0, S0, 128, 0, WME, WPE, BM, BP, 64, 128, true, true,
            SD(4, 3), SD(5, 2), grow0, trow, nullptr, lane);
  __syncthreads();
  stage_weights(P.fw[5], P.fr[5], P.fmb[5], P.frb[5], 128, 64, true, SD(5, 0), WME, WPE, BM, BP);
  __syncthreads();
  run_layer(B0, S0, 128, B1, nullptr, 128, 64, WME, WPE, BM, BP, 128, 64, true, true,
            SD(5, 3), 0u, grow0, trow, nullptr, lane);   // -> merge cols 64..127 in B1

  // ---- positional encoding add (wave-private rows; index hoisted per row) --
#pragma unroll 1
  for (int r = 0; r < 16; ++r) {
    int grow = grow0 + r;
    float f = P.sample_freq[grow];
    int idx = (f > 40.f) ? 0 : (f > 20.f) ? 1 : (f > 7.5f) ? 2 : 3;
    const float* perow = P.pe + idx * 128;
    _Float16* mrow = B1 + (trow + r) * 128;
#pragma unroll
    for (int c = lane; c < 128; c += 32)
      mrow[c] = (_Float16)((float)mrow[c] + perow[c]);
  }
  __syncthreads();

  // ---- m1 (dense 128->128), m2 (dense 128->64, feat_map out) ----
  stage_weights(P.m1w, nullptr, P.m1b, nullptr, 128, 128, false, 0u, WME, WPE, BM, BP);
  __syncthreads();
  run_layer(B1, nullptr, 128, B0, nullptr, 128, 0, WME, WPE, BM, BP, 128, 128, false, false,
            0u, 0u, grow0, trow, nullptr, lane);
  __syncthreads();
  stage_weights(P.m2w, nullptr, P.m2b, nullptr, 128, 64, false, 0u, WME, WPE, BM, BP);
  __syncthreads();
  run_layer(B0, nullptr, 128, B2, S0, 64, 0, WME, WPE, BM, BP, 128, 64, false, false,
            0u, SD(6, 2), grow0, trow, P.out_feat, lane);  // feat_map + signed for b4
  __syncthreads();

  // ---- b4 (flipout 64->32), b5 (flipout 32->7) + residual state out ----
  stage_weights(P.fw[6], P.fr[6], P.fmb[6], P.frb[6], 64, 32, true, SD(6, 0), WME, WPE, BM, BP);
  __syncthreads();
  run_layer(B2, S0, 64, B0, S1, 32, 0, WME, WPE, BM, BP, 64, 32, true, false,
            SD(6, 3), SD(7, 2), grow0, trow, nullptr, lane);
  __syncthreads();
  stage_weights(P.fw[7], P.fr[7], P.fmb[7], P.frb[7], 32, 7, true, SD(7, 0), WME, WPE, BM, BP);
  __syncthreads();
  {
    v8f acc = {}, acc2 = {};
    v16h a  = loadA(B0, 32, trow, 0, lane);
    v16h b  = loadB(WME, 1, 0, 0, lane);
    acc = __builtin_amdgcn_wmma_f32_16x16x32_f16(false, a, false, b, (short)0, acc, false, false);
    v16h a2 = loadA(S1, 32, trow, 0, lane);
    v16h b2 = loadB(WPE, 1, 0, 0, lane);
    acc2 = __builtin_amdgcn_wmma_f32_16x16x32_f16(false, a2, false, b2, (short)0, acc2, false, false);
    const int n = lane & 15;
    const int mh = (lane >> 4) * 8;
#pragma unroll
    for (int r = 0; r < 8; ++r) {
      int grow = grow0 + mh + r;
      if (n < 7) {
        float upd = acc[r] + BM[n] +
                    (acc2[r] + BP[n]) * bitsgn(mix1(SD(7, 3), (unsigned)grow, (unsigned)n), 0);
        P.out_state[grow * 7 + n] = P.last_state[grow * 7 + n] + upd;
      }
    }
  }
}

extern "C" void kernel_launch(void* const* d_in, const int* in_sizes, int n_in,
                              void* d_out, int out_size, void* d_ws, size_t ws_size,
                              hipStream_t stream) {
  (void)in_sizes; (void)n_in; (void)out_size; (void)d_ws; (void)ws_size;
  Params P;
  P.last_state  = (const float*)d_in[0];
  P.action      = (const float*)d_in[1];
  P.sample_freq = (const float*)d_in[2];
  P.pe          = (const float*)d_in[3];
  for (int l = 0; l < 8; ++l) {
    P.fw[l]  = (const float*)d_in[4 + l * 4 + 0];
    P.fr[l]  = (const float*)d_in[4 + l * 4 + 1];
    P.fmb[l] = (const float*)d_in[4 + l * 4 + 2];
    P.frb[l] = (const float*)d_in[4 + l * 4 + 3];
  }
  P.m1w = (const float*)d_in[36];
  P.m1b = (const float*)d_in[37];
  P.m2w = (const float*)d_in[38];
  P.m2b = (const float*)d_in[39];
  P.out_state = (float*)d_out;
  P.out_feat  = (float*)d_out + (size_t)N_ROWS * 7;

  const size_t shmem = (size_t)(8192 * 5 + 16384 + 8192) * 2 + 256 * 4;  // 132096 B
  dim3 grid(N_ROWS / ROWS_PB), block(THREADS);
  flipout_mlp_kernel<<<grid, block, shmem, stream>>>(P);
}
